// Nonlocal_correlation_91328184582695
// MI455X (gfx1250) — matrix operand order repaired and checked by reading, not compiled, not run on hardware
//
#include <hip/hip_runtime.h>

typedef __attribute__((ext_vector_type(16))) __bf16 v16bf;
typedef __attribute__((ext_vector_type(8)))  float  v8f;
typedef unsigned int u32;

// ---------------------------------------------------------------------------
// WMMA fragment helpers (CDNA5 wave32; k = 16*(lane/16) + s in A and B, not the ISA table order)
// ---------------------------------------------------------------------------

// A-matrix 16x32 bf16, row-major source A[m][k] = base[m*stride + k]
// (row base must be 16B aligned for b128 merging: stride*2 % 16 == 0)
__device__ __forceinline__ v16bf load_a_frag(const __bf16* base, int stride, int lane) {
    const int hf = lane >> 4, m = lane & 15;
    const __bf16* row = base + m * stride;
    v16bf f;
#pragma unroll
    for (int v = 0; v < 8; ++v) {
        int k0 = 16 * hf + 2 * v;
        f[2 * v]     = row[k0];
        f[2 * v + 1] = row[k0 + 1];
    }
    return f;
}

// B-matrix 32x16 bf16 where the logical B[k][n] = base[n*stride + k]  (B = tile^T)
__device__ __forceinline__ v16bf load_bT_frag(const __bf16* base, int stride, int lane) {
    const int hf = lane >> 4, n = lane & 15;
    const __bf16* row = base + n * stride;
    v16bf f;
#pragma unroll
    for (int v = 0; v < 8; ++v) {
        int k = 16 * hf + 2 * v;
        f[2 * v]     = row[k];
        f[2 * v + 1] = row[k + 1];
    }
    return f;
}

// B-matrix 32x16 bf16 from K-pair-interleaved storage:
// element (k,n) lives at dword [( (k>>1) * strideU ) + n], low half = even k.
__device__ __forceinline__ v16bf load_bP_frag(const u32* base, int strideU, int lane) {
    const int hf = lane >> 4, n = lane & 15;
    v16bf f;
#pragma unroll
    for (int v = 0; v < 8; ++v) {
        union { u32 u; __bf16 h[2]; } cv;
        cv.u = base[(8 * hf + v) * strideU + n];
        f[2 * v]     = cv.h[0];
        f[2 * v + 1] = cv.h[1];
    }
    return f;
}

// ---------------------------------------------------------------------------
#define BB    8
#define CIN   768
#define NN    4096
#define MID   256
#define INTER 128
#define OUTC  20
#define BN_EPS 1e-5f

// ---------------------------------------------------------------------------
// K0: fold weights.
//   WtP_z : (w_z @ w_conv) stored K-pair-interleaved over c: ((c>>1)*128+i)*2+(c&1)
//   W2    = w_out @ w_conv                     [20][768] f32
//   W1    = w_out @ (inv*w_W)                  [20][128] f32
//   cvec  = w_out @ (inv*b_W + beta - mean*inv) + b_out
// ---------------------------------------------------------------------------
__global__ __launch_bounds__(256) void fuse_weights(
        const float* __restrict__ w_conv, const float* __restrict__ w_theta,
        const float* __restrict__ w_phi,  const float* __restrict__ w_g,
        const float* __restrict__ w_W,    const float* __restrict__ b_W,
        const float* __restrict__ gamma,  const float* __restrict__ beta,
        const float* __restrict__ mean,   const float* __restrict__ var,
        const float* __restrict__ w_out,  const float* __restrict__ b_out,
        __bf16* Wt0, __bf16* Wt1, __bf16* Wt2,
        float* W1, float* W2, float* cvec) {
    int idx = blockIdx.x * 256 + threadIdx.x;
    const int JA = 3 * INTER * CIN;
    if (idx < JA) {
        int z = idx / (INTER * CIN);
        int r = idx % (INTER * CIN);
        int i = r / CIN, k = r % CIN;
        const float* wz = (z == 0) ? w_theta : (z == 1) ? w_phi : w_g;
        float s = 0.f;
        for (int m = 0; m < MID; ++m) s += wz[i * MID + m] * w_conv[m * CIN + k];
        __bf16* Wt = (z == 0) ? Wt0 : (z == 1) ? Wt1 : Wt2;
        Wt[((size_t)(k >> 1) * INTER + i) * 2 + (k & 1)] = (__bf16)s;  // pair-interleaved
        return;
    }
    idx -= JA;
    if (idx < OUTC * CIN) {
        int o = idx / CIN, k = idx % CIN;
        float s = 0.f;
        for (int m = 0; m < MID; ++m) s += w_out[o * MID + m] * w_conv[m * CIN + k];
        W2[o * CIN + k] = s;
        return;
    }
    idx -= OUTC * CIN;
    if (idx < OUTC * INTER) {
        int o = idx / INTER, i = idx % INTER;
        float s = 0.f;
        for (int m = 0; m < MID; ++m) {
            float inv = gamma[m] * rsqrtf(var[m] + BN_EPS);
            s += w_out[o * MID + m] * inv * w_W[m * INTER + i];
        }
        W1[o * INTER + i] = s;
        return;
    }
    idx -= OUTC * INTER;
    if (idx < OUTC) {
        int o = idx;
        float s = b_out[o];
        for (int m = 0; m < MID; ++m) {
            float inv = gamma[m] * rsqrtf(var[m] + BN_EPS);
            s += w_out[o * MID + m] * (inv * b_W[m] + beta[m] - mean[m] * inv);
        }
        cvec[o] = s;
    }
}

// ---------------------------------------------------------------------------
// K1: Q/K/V projections.  out[b][n][i] = sum_c A[i][c] * x[b][c][n]
// GEMM: M = 64 spatial rows/block, N = 128 channels, K = 768.
// grid = (N/64, B, 3); 8 waves: wave w owns channel strip nt=w, mt=0..3.
// Q,K stored [b][n][i] bf16;  V stored K-pair-interleaved over n:
//   ((b*2048 + (n>>1))*128 + i)*2 + (n&1)
// ---------------------------------------------------------------------------
__global__ __launch_bounds__(256) void qkv_kernel(
        const float* __restrict__ x1, const float* __restrict__ x2,
        const __bf16* __restrict__ Wt0, const __bf16* __restrict__ Wt1,
        const __bf16* __restrict__ Wt2,
        __bf16* Qg, __bf16* Kg, __bf16* Vg) {
    __shared__ __bf16 Xl[64 * 40];    // A tile: 64 spatial x 32 k  (80B rows)
    __shared__ u32    WlU[16 * 136];  // B tile: 16 k-pairs x 128 ch (544B rows)

    const int tid = threadIdx.x, lane = tid & 31, w = tid >> 5;
    const int hf = lane >> 4, ln = lane & 15;
    const int n0 = blockIdx.x * 64;
    const int b  = blockIdx.y;
    const int z  = blockIdx.z;
    const float*  xs  = (z == 0) ? x1  : x2;
    const __bf16* Wt  = (z == 0) ? Wt0 : (z == 1) ? Wt1 : Wt2;
    __bf16*       dst = (z == 0) ? Qg  : (z == 1) ? Kg  : Vg;
    const float* xb = xs + (size_t)b * CIN * NN;

    v8f acc[4];
#pragma unroll
    for (int t = 0; t < 4; ++t)
#pragma unroll
        for (int r = 0; r < 8; ++r) acc[t][r] = 0.f;

    for (int k0 = 0; k0 < CIN; k0 += 32) {
        // stage weight tile (already pair-interleaved in global): uint4 copy
        const uint4* wsrc = (const uint4*)(Wt + (size_t)k0 * INTER);
        for (int c = tid; c < 512; c += 256) {
            int pr = c >> 5, off = c & 31;
            *(uint4*)(WlU + pr * 136 + off * 4) = wsrc[c];
        }
        // stage x tile transposed: each thread converts 4 k for one nn -> b64 store
        for (int c = tid; c < 512; c += 256) {
            int nn = c & 63, kq = c >> 6;  // kq = 0..7 (4 k each)
            const float* xp = xb + (size_t)(k0 + kq * 4) * NN + n0 + nn;
            union { unsigned long long u; __bf16 h[4]; } cv;
            cv.h[0] = (__bf16)xp[0];
            cv.h[1] = (__bf16)xp[NN];
            cv.h[2] = (__bf16)xp[2 * NN];
            cv.h[3] = (__bf16)xp[3 * NN];
            *(unsigned long long*)(Xl + nn * 40 + kq * 4) = cv.u;
        }
        __syncthreads();
        v16bf bf = load_bP_frag(WlU + 16 * w, 136, lane);
#pragma unroll
        for (int mt = 0; mt < 4; ++mt) {
            v16bf af = load_a_frag(Xl + mt * 16 * 40, 40, lane);
            acc[mt] = __builtin_amdgcn_wmma_f32_16x16x32_bf16(
                false, af, false, bf, (short)0, acc[mt], false, false);
        }
        __syncthreads();
    }

    const int icol = 16 * w + ln;
    if (z == 2) {  // V: K-pair-interleaved over spatial index
#pragma unroll
        for (int mt = 0; mt < 4; ++mt)
#pragma unroll
            for (int r = 0; r < 8; ++r) {
                int q = n0 + mt * 16 + r + 8 * hf;
                dst[(((size_t)b * (NN / 2) + (q >> 1)) * INTER + icol) * 2 + (q & 1)] =
                    (__bf16)acc[mt][r];
            }
    } else {
#pragma unroll
        for (int mt = 0; mt < 4; ++mt)
#pragma unroll
            for (int r = 0; r < 8; ++r) {
                int q = n0 + mt * 16 + r + 8 * hf;
                dst[((size_t)b * NN + q) * INTER + icol] = (__bf16)acc[mt][r];
            }
    }
}

// ---------------------------------------------------------------------------
// K2: flash attention.  grid = (N/64, B), 256 threads (8 waves).
// ---------------------------------------------------------------------------
__global__ __launch_bounds__(256) void attn_kernel(
        const __bf16* __restrict__ Qg, const __bf16* __restrict__ Kg,
        const __bf16* __restrict__ Vg, __bf16* __restrict__ Yg) {
    extern __shared__ char smem[];
    __bf16* Qs = (__bf16*)smem;            // 64 x 136  (272B rows, 16B aligned)
    __bf16* Ks = Qs + 64 * 136;            // 64 x 136
    u32*    Vu = (u32*)(Ks + 64 * 136);    // 32 pair-rows x 136 dwords (544B rows)
    float*  Ss = (float*)(Vu + 32 * 136);  // 64 x 68 f32
    __bf16* Ps = (__bf16*)(Ss + 64 * 68);  // 64 x 72  (144B rows)
    float*  rmax   = (float*)(Ps + 64 * 72);
    float*  rsum   = rmax + 64;
    float*  rscale = rsum + 64;

    const int tid = threadIdx.x, lane = tid & 31, w = tid >> 5;
    const int hf = lane >> 4, ln = lane & 15;
    const int b  = blockIdx.y;
    const int n0 = blockIdx.x * 64;

    const __bf16* Qb = Qg + ((size_t)b * NN + n0) * INTER;
    const __bf16* Kb = Kg + (size_t)b * NN * INTER;
    const __bf16* Vb = Vg + (size_t)b * NN * INTER;   // pair-interleaved

    // stage Q block: uint4 copy (8 bf16 per inst)
    {
        const uint4* src = (const uint4*)Qb;
        for (int c = tid; c < 1024; c += 256) {
            int q = c >> 4, off = c & 15;
            *(uint4*)(Qs + q * 136 + off * 8) = src[c];
        }
    }
    if (tid < 64) { rmax[tid] = -3.0e38f; rsum[tid] = 0.f; }

    v8f acc[4];   // wave w owns channel strip nt=w, query tiles mt=0..3
#pragma unroll
    for (int t = 0; t < 4; ++t)
#pragma unroll
        for (int r = 0; r < 8; ++r) acc[t][r] = 0.f;
    __syncthreads();

    for (int kv0 = 0; kv0 < NN; kv0 += 64) {
        // stage K (row-major) and V (pair-interleaved) tiles: uint4 copies
        {
            const uint4* ksrc = (const uint4*)(Kb + (size_t)kv0 * INTER);
            const uint4* vsrc = (const uint4*)(Vb + (size_t)kv0 * INTER);
            for (int c = tid; c < 1024; c += 256) {
                int kr = c >> 4, koff = c & 15;
                *(uint4*)(Ks + kr * 136 + koff * 8) = ksrc[c];
                int vp = c >> 5, voff = c & 31;
                *(uint4*)(Vu + vp * 136 + voff * 4) = vsrc[c];
            }
        }
        if (kv0 + 64 < NN) {
            __builtin_prefetch(Kb + (size_t)(kv0 + 64) * INTER + tid * 32, 0, 0);
            __builtin_prefetch(Vb + (size_t)(kv0 + 64) * INTER + tid * 32, 0, 0);
        }
        __syncthreads();

        // S = Q @ K^T : 16 tiles of 16x16, 2 per wave
#pragma unroll
        for (int tt = 0; tt < 2; ++tt) {
            int t = 2 * w + tt, mt = t >> 2, nt = t & 3;
            v8f s;
#pragma unroll
            for (int r = 0; r < 8; ++r) s[r] = 0.f;
#pragma unroll
            for (int ks = 0; ks < 4; ++ks) {
                v16bf af = load_a_frag (Qs + mt * 16 * 136 + 32 * ks, 136, lane);
                v16bf bf = load_bT_frag(Ks + nt * 16 * 136 + 32 * ks, 136, lane);
                s = __builtin_amdgcn_wmma_f32_16x16x32_bf16(
                    false, af, false, bf, (short)0, s, false, false);
            }
#pragma unroll
            for (int r = 0; r < 8; ++r)
                Ss[(mt * 16 + r + 8 * hf) * 68 + nt * 16 + ln] = s[r];
        }
        __syncthreads();

        // online softmax: 4 lanes per row, 16 cols each
        {
            int row = tid >> 2, quad = tid & 3;
            const float* srow = &Ss[row * 68 + quad * 16];
            float mx = -3.0e38f;
#pragma unroll
            for (int c = 0; c < 16; ++c) mx = fmaxf(mx, srow[c]);
            mx = fmaxf(mx, __shfl_xor(mx, 1));
            mx = fmaxf(mx, __shfl_xor(mx, 2));
            float mold = rmax[row];
            float mnew = fmaxf(mold, mx);
            float sum = 0.f;
            u32* prow = (u32*)(Ps + row * 72 + quad * 16);
#pragma unroll
            for (int c = 0; c < 16; c += 2) {
                float p0 = __expf(srow[c] - mnew);
                float p1 = __expf(srow[c + 1] - mnew);
                sum += p0 + p1;
                union { u32 u; __bf16 h[2]; } cv;
                cv.h[0] = (__bf16)p0;
                cv.h[1] = (__bf16)p1;
                prow[c >> 1] = cv.u;
            }
            sum += __shfl_xor(sum, 1);
            sum += __shfl_xor(sum, 2);
            if (quad == 0) {
                float sc = __expf(mold - mnew);
                rscale[row] = sc;
                rsum[row] = rsum[row] * sc + sum;
                rmax[row] = mnew;
            }
        }
        __syncthreads();

        // rescale accumulators, then O += P @ V
#pragma unroll
        for (int mt = 0; mt < 4; ++mt)
#pragma unroll
            for (int r = 0; r < 8; ++r)
                acc[mt][r] *= rscale[mt * 16 + r + 8 * hf];
#pragma unroll
        for (int ks = 0; ks < 2; ++ks) {
            v16bf bf = load_bP_frag(Vu + 16 * ks * 136 + 16 * w, 136, lane);
#pragma unroll
            for (int mt = 0; mt < 4; ++mt) {
                v16bf af = load_a_frag(Ps + mt * 16 * 72 + 32 * ks, 72, lane);
                acc[mt] = __builtin_amdgcn_wmma_f32_16x16x32_bf16(
                    false, af, false, bf, (short)0, acc[mt], false, false);
            }
        }
        __syncthreads();
    }

    const int icol = 16 * w + ln;
#pragma unroll
    for (int mt = 0; mt < 4; ++mt)
#pragma unroll
        for (int r = 0; r < 8; ++r) {
            int q = mt * 16 + r + 8 * hf;
            float v = acc[mt][r] / rsum[q];
            Yg[((size_t)b * NN + n0 + q) * INTER + icol] = (__bf16)v;
        }
}

// ---------------------------------------------------------------------------
// K3: epilogue.  out[b][o][n] = cvec[o] + W2[o][:]·x1[b][:][n] + W1[o][:]·y[b][n][:]
// ---------------------------------------------------------------------------
__global__ __launch_bounds__(256) void out_kernel(
        const float* __restrict__ x1, const __bf16* __restrict__ Yg,
        const float* __restrict__ W1, const float* __restrict__ W2,
        const float* __restrict__ cvec, float* __restrict__ out) {
    const int n = blockIdx.x * 256 + threadIdx.x;
    const int b = blockIdx.y;
    float acc[OUTC];
#pragma unroll
    for (int o = 0; o < OUTC; ++o) acc[o] = cvec[o];

    const float* xb = x1 + (size_t)b * CIN * NN + n;
    for (int c = 0; c < CIN; ++c) {
        float xv = xb[(size_t)c * NN];
#pragma unroll
        for (int o = 0; o < OUTC; ++o) acc[o] += W2[o * CIN + c] * xv;
    }
    const __bf16* yb = Yg + ((size_t)b * NN + n) * INTER;
    for (int i = 0; i < INTER; ++i) {
        float yv = (float)yb[i];
#pragma unroll
        for (int o = 0; o < OUTC; ++o) acc[o] += W1[o * INTER + i] * yv;
    }
#pragma unroll
    for (int o = 0; o < OUTC; ++o)
        out[((size_t)(b * OUTC + o)) * NN + n] = acc[o];
}

// ---------------------------------------------------------------------------
extern "C" void kernel_launch(void* const* d_in, const int* in_sizes, int n_in,
                              void* d_out, int out_size, void* d_ws, size_t ws_size,
                              hipStream_t stream) {
    (void)in_sizes; (void)n_in; (void)out_size; (void)ws_size;
    const float* x1      = (const float*)d_in[0];
    const float* x2      = (const float*)d_in[1];
    const float* w_conv  = (const float*)d_in[2];
    const float* w_theta = (const float*)d_in[3];
    const float* w_phi   = (const float*)d_in[4];
    const float* w_g     = (const float*)d_in[5];
    const float* w_W     = (const float*)d_in[6];
    const float* b_W     = (const float*)d_in[7];
    const float* bn_g    = (const float*)d_in[8];
    const float* bn_b    = (const float*)d_in[9];
    const float* bn_m    = (const float*)d_in[10];
    const float* bn_v    = (const float*)d_in[11];
    const float* w_out   = (const float*)d_in[12];
    const float* b_out   = (const float*)d_in[13];
    float* out = (float*)d_out;

    char* ws = (char*)d_ws;
    const size_t SZ_WT = (size_t)CIN * INTER * 2;          // 196608 B (bf16)
    __bf16* Wt0 = (__bf16*)(ws);
    __bf16* Wt1 = (__bf16*)(ws + SZ_WT);
    __bf16* Wt2 = (__bf16*)(ws + 2 * SZ_WT);
    float* W1   = (float*)(ws + 3 * SZ_WT);
    float* W2   = W1 + OUTC * INTER;
    float* cvec = W2 + OUTC * CIN;
    size_t off = 3 * SZ_WT + (size_t)(OUTC * INTER + OUTC * CIN + OUTC) * 4;
    off = (off + 255) & ~(size_t)255;
    const size_t SZ_M = (size_t)BB * NN * INTER * 2;       // 8 MB each (bf16)
    __bf16* Qg = (__bf16*)(ws + off);
    __bf16* Kg = (__bf16*)(ws + off + SZ_M);
    __bf16* Vg = (__bf16*)(ws + off + 2 * SZ_M);
    __bf16* Yg = (__bf16*)(ws + off + 3 * SZ_M);

    const int T = 3 * INTER * CIN + OUTC * CIN + OUTC * INTER + OUTC;
    fuse_weights<<<dim3((T + 255) / 256), dim3(256), 0, stream>>>(
        w_conv, w_theta, w_phi, w_g, w_W, b_W, bn_g, bn_b, bn_m, bn_v,
        w_out, b_out, Wt0, Wt1, Wt2, W1, W2, cvec);

    qkv_kernel<<<dim3(NN / 64, BB, 3), dim3(256), 0, stream>>>(
        x1, x2, Wt0, Wt1, Wt2, Qg, Kg, Vg);

    const size_t ATTN_LDS = (size_t)(2 * 64 * 136 * 2 + 32 * 136 * 4 +
                                     64 * 68 * 4 + 64 * 72 * 2 + 3 * 64 * 4);
    attn_kernel<<<dim3(NN / 64, BB), dim3(256), ATTN_LDS, stream>>>(Qg, Kg, Vg, Yg);

    out_kernel<<<dim3(NN / 256, BB), dim3(256), 0, stream>>>(x1, Yg, W1, W2, cvec, out);
}
